// LaCTEnergyAwareTTTRouter_5059471475441
// MI455X (gfx1250) — matrix operand order repaired and checked by reading, not compiled, not run on hardware
//
#include <hip/hip_runtime.h>
#include <math.h>

// ---------------------------------------------------------------------------
// Problem constants (from reference)
// ---------------------------------------------------------------------------
#define D_MODEL   2048
#define HIDDEN    4096
#define N_EXP     64
#define N_TOKENS  16384
#define LN_EPS    1e-5f

#define TILE_M    128    // tokens per block (16 wave32 waves)
#define NCH       64     // hidden chunk width fused into router GEMM

typedef __bf16 bf16_t;
typedef __attribute__((ext_vector_type(16))) __bf16 v16bf;
typedef __attribute__((ext_vector_type(8)))  float  v8f;

union ABfrag {
    v16bf v;
    uint4 q[2];
};

union Pack4 {
    bf16_t b[4];
    uint2  u;
};

__device__ __forceinline__ v8f vzero8() {
    v8f v;
#pragma unroll
    for (int i = 0; i < 8; ++i) v[i] = 0.0f;
    return v;
}

// ---------------------------------------------------------------------------
// Kernel 1: fold ln_g into w1/w2 columns, convert w1/w2/w3 to bf16,
//           fold b3+expert_bias, zero the usage histogram.
// ---------------------------------------------------------------------------
__global__ void prep_weights_kernel(const float* __restrict__ w1,
                                    const float* __restrict__ w2,
                                    const float* __restrict__ w3,
                                    const float* __restrict__ ln_g,
                                    const float* __restrict__ b3,
                                    const float* __restrict__ ebias,
                                    bf16_t* __restrict__ w1g,
                                    bf16_t* __restrict__ w2g,
                                    bf16_t* __restrict__ w3b,
                                    float*  __restrict__ b3f,
                                    float*  __restrict__ usage) {
    const size_t stride = (size_t)gridDim.x * blockDim.x;
    const size_t i0 = (size_t)blockIdx.x * blockDim.x + threadIdx.x;

    for (size_t j = i0; j < (size_t)HIDDEN * D_MODEL; j += stride) {
        const float g = ln_g[j & (D_MODEL - 1)];
        w1g[j] = (bf16_t)(w1[j] * g);
        w2g[j] = (bf16_t)(w2[j] * g);
    }
    for (size_t j = i0; j < (size_t)N_EXP * HIDDEN; j += stride) {
        w3b[j] = (bf16_t)w3[j];
    }
    if (i0 < N_EXP) {
        b3f[i0]   = b3[i0] + ebias[i0];
        usage[i0] = 0.0f;
    }
}

// ---------------------------------------------------------------------------
// Kernel 2: fold ln_b through w1/w2 into the hidden biases.
// One wave (32 lanes) per hidden row; 8192 rows total.
// ---------------------------------------------------------------------------
__global__ void prep_bias_kernel(const float* __restrict__ w1,
                                 const float* __restrict__ w2,
                                 const float* __restrict__ ln_b,
                                 const float* __restrict__ b1,
                                 const float* __restrict__ b2,
                                 float* __restrict__ b1f,
                                 float* __restrict__ b2f) {
    const int gw   = (int)(((size_t)blockIdx.x * blockDim.x + threadIdx.x) >> 5);
    const int lane = threadIdx.x & 31;
    if (gw >= 2 * HIDDEN) return;

    const bool first = (gw < HIDDEN);
    const int  rowi  = gw & (HIDDEN - 1);
    const float* row = (first ? w1 : w2) + (size_t)rowi * D_MODEL;

    float s = 0.0f;
    for (int k = lane * 4; k < D_MODEL; k += 128) {
        const float4 wv = *(const float4*)(row + k);
        const float4 bv = *(const float4*)(ln_b + k);
        s += wv.x * bv.x + wv.y * bv.y + wv.z * bv.z + wv.w * bv.w;
    }
#pragma unroll
    for (int off = 16; off > 0; off >>= 1) s += __shfl_xor(s, off, 32);
    if (lane == 0) {
        if (first) b1f[rowi] = b1[rowi] + s;
        else       b2f[rowi] = b2[rowi] + s;
    }
}

// ---------------------------------------------------------------------------
// Kernel 3: LayerNorm (stats + normalize) to bf16, one wave per token.
// Row is held in registers (64 floats/lane), x is read exactly once.
// xn_stored = (x - mu) * rsqrt(var + eps); ln_g/ln_b already folded away.
// ---------------------------------------------------------------------------
__global__ void ln_norm_kernel(const float* __restrict__ x,
                               bf16_t* __restrict__ xnb) {
    const int wave  = threadIdx.x >> 5;
    const int lane  = threadIdx.x & 31;
    const int token = blockIdx.x * 8 + wave;
    if (token >= N_TOKENS) return;

    const float* row = x + (size_t)token * D_MODEL;
    float4 v[16];
    float s = 0.0f, s2 = 0.0f;
#pragma unroll
    for (int i = 0; i < 16; ++i) {
        v[i] = *(const float4*)(row + lane * 4 + i * 128);
        s  += v[i].x + v[i].y + v[i].z + v[i].w;
        s2 += v[i].x * v[i].x + v[i].y * v[i].y + v[i].z * v[i].z + v[i].w * v[i].w;
    }
#pragma unroll
    for (int off = 16; off > 0; off >>= 1) {
        s  += __shfl_xor(s,  off, 32);
        s2 += __shfl_xor(s2, off, 32);
    }
    const float mu  = s * (1.0f / D_MODEL);
    const float var = s2 * (1.0f / D_MODEL) - mu * mu;
    const float rs  = rsqrtf(var + LN_EPS);

    bf16_t* orow = xnb + (size_t)token * D_MODEL;
#pragma unroll
    for (int i = 0; i < 16; ++i) {
        Pack4 pk;
        pk.b[0] = (bf16_t)((v[i].x - mu) * rs);
        pk.b[1] = (bf16_t)((v[i].y - mu) * rs);
        pk.b[2] = (bf16_t)((v[i].z - mu) * rs);
        pk.b[3] = (bf16_t)((v[i].w - mu) * rs);
        *(uint2*)(orow + lane * 4 + i * 128) = pk.u;
    }
}

// ---------------------------------------------------------------------------
// Kernel 4: fused (w1,w2) GEMMs -> SiLU-gate -> w3 GEMM -> softmax/top2.
// 128-token tile per block, 16 wave32 waves, bf16 WMMA with f32 accumulate.
// Wave w: token-frag row r = w>>1 (rows r*16..r*16+15, r in 0..7),
//         column-pair  p = w&1  (hidden cols p*32..p*32+31 within a chunk;
//                                also expert cols p*32..p*32+31 for logits).
// ---------------------------------------------------------------------------
__global__ __launch_bounds__(512, 1) void moe_kernel(
        const bf16_t* __restrict__ xnb,
        const bf16_t* __restrict__ w1g,
        const bf16_t* __restrict__ w2g,
        const bf16_t* __restrict__ w3b,
        const float*  __restrict__ b1f,
        const float*  __restrict__ b2f,
        const float*  __restrict__ b3f,
        float* __restrict__ out_idx,
        float* __restrict__ out_p,
        float* __restrict__ usage) {

    __shared__ __align__(16) bf16_t s_sw[2][TILE_M][NCH + 8];  // 36.0 KB
    __shared__ float s_logit[TILE_M][N_EXP + 1];               // 32.5 KB

    const int tid    = threadIdx.x;
    const int wave   = tid >> 5;
    const int lane   = tid & 31;
    const int lane_m = lane & 15;
    const int kg     = lane >> 4;          // lane group 0/1
    const int r      = wave >> 1;          // token-frag row 0..7
    const int p      = wave & 1;           // column-pair 0..1
    const int m0     = blockIdx.x * TILE_M;
    const int row    = m0 + r * 16 + lane_m;

    const bf16_t* xr = xnb + (size_t)row * D_MODEL;

    v8f lacc[2];
    lacc[0] = vzero8();
    lacc[1] = vzero8();

    for (int n0 = 0; n0 < HIDDEN; n0 += NCH) {
        const int buf = (n0 / NCH) & 1;
        const int cb0 = n0 + p * 32;       // this wave's hidden-col base

        v8f acc1[2], acc2[2];
        acc1[0] = vzero8(); acc1[1] = vzero8();
        acc2[0] = vzero8(); acc2[1] = vzero8();

        // ---- GEMM1/GEMM2: h1,h2 tile over full K = d_model ----
        for (int k0 = 0; k0 < D_MODEL; k0 += 32) {
            // A fragment: pre-normalized bf16, two b128 loads
            const int kb = k0 + kg * 16;
            ABfrag A;
            A.q[0] = *(const uint4*)(xr + kb);
            A.q[1] = *(const uint4*)(xr + kb + 8);

            // B fragments: rows of w1g/w2g give the K-packed per-lane layout
            const int kbB = k0 + kg * 16;
            ABfrag B1a, B1b, B2a, B2b;
            {
                const bf16_t* p1 = w1g + (size_t)(cb0 + lane_m) * D_MODEL + kbB;
                B1a.q[0] = *(const uint4*)(p1);
                B1a.q[1] = *(const uint4*)(p1 + 8);
                const bf16_t* p1b = p1 + (size_t)16 * D_MODEL;
                B1b.q[0] = *(const uint4*)(p1b);
                B1b.q[1] = *(const uint4*)(p1b + 8);
                const bf16_t* p2 = w2g + (size_t)(cb0 + lane_m) * D_MODEL + kbB;
                B2a.q[0] = *(const uint4*)(p2);
                B2a.q[1] = *(const uint4*)(p2 + 8);
                const bf16_t* p2b = p2 + (size_t)16 * D_MODEL;
                B2b.q[0] = *(const uint4*)(p2b);
                B2b.q[1] = *(const uint4*)(p2b + 8);
            }
            acc1[0] = __builtin_amdgcn_wmma_f32_16x16x32_bf16(
                false, A.v, false, B1a.v, (short)0, acc1[0], false, false);
            acc1[1] = __builtin_amdgcn_wmma_f32_16x16x32_bf16(
                false, A.v, false, B1b.v, (short)0, acc1[1], false, false);
            acc2[0] = __builtin_amdgcn_wmma_f32_16x16x32_bf16(
                false, A.v, false, B2a.v, (short)0, acc2[0], false, false);
            acc2[1] = __builtin_amdgcn_wmma_f32_16x16x32_bf16(
                false, A.v, false, B2b.v, (short)0, acc2[1], false, false);
        }

        // ---- bias + SiLU gate, stage swish chunk into LDS (bf16) ----
        {
            float bb1[2], bb2[2];
            bb1[0] = b1f[cb0 + lane_m];      bb1[1] = b1f[cb0 + 16 + lane_m];
            bb2[0] = b2f[cb0 + lane_m];      bb2[1] = b2f[cb0 + 16 + lane_m];
#pragma unroll
            for (int nf = 0; nf < 2; ++nf) {
#pragma unroll
                for (int e = 0; e < 8; ++e) {
                    const int trow = r * 16 + e + kg * 8;     // C layout: M = e + kg*8
                    const int tcol = p * 32 + nf * 16 + lane_m;
                    const float h1 = acc1[nf][e] + bb1[nf];
                    const float h2 = acc2[nf][e] + bb2[nf];
                    const float sw = h1 / (1.0f + __expf(-h2));
                    s_sw[buf][trow][tcol] = (bf16_t)sw;
                }
            }
        }
        __syncthreads();

        // ---- router GEMM: logits += swish_chunk @ w3_chunk^T ----
#pragma unroll
        for (int ks = 0; ks < NCH; ks += 32) {
            ABfrag A2;
            const int kb2 = ks + kg * 16;
            const bf16_t* sp = &s_sw[buf][r * 16 + lane_m][kb2];
            A2.q[0] = *(const uint4*)(sp);
            A2.q[1] = *(const uint4*)(sp + 8);

            ABfrag B3a, B3b;
            const int kbB2 = n0 + ks + kg * 16;
            const bf16_t* p3 = w3b + (size_t)(p * 32 + lane_m) * HIDDEN + kbB2;
            B3a.q[0] = *(const uint4*)(p3);
            B3a.q[1] = *(const uint4*)(p3 + 8);
            const bf16_t* p3b = p3 + (size_t)16 * HIDDEN;
            B3b.q[0] = *(const uint4*)(p3b);
            B3b.q[1] = *(const uint4*)(p3b + 8);

            lacc[0] = __builtin_amdgcn_wmma_f32_16x16x32_bf16(
                false, A2.v, false, B3a.v, (short)0, lacc[0], false, false);
            lacc[1] = __builtin_amdgcn_wmma_f32_16x16x32_bf16(
                false, A2.v, false, B3b.v, (short)0, lacc[1], false, false);
        }
        // double-buffered s_sw: the single per-chunk barrier is sufficient
    }

    // ---- logits (+ b3 + expert_bias) to LDS ----
#pragma unroll
    for (int nf = 0; nf < 2; ++nf) {
        const int ecol = p * 32 + nf * 16 + lane_m;
        const float b3v = b3f[ecol];
#pragma unroll
        for (int e = 0; e < 8; ++e) {
            s_logit[r * 16 + e + kg * 8][ecol] = lacc[nf][e] + b3v;
        }
    }
    __syncthreads();

    // ---- softmax + top-2 per token ----
    if (tid < TILE_M) {
        const float* lg = s_logit[tid];
        float mx = lg[0];
#pragma unroll 8
        for (int e = 1; e < N_EXP; ++e) mx = fmaxf(mx, lg[e]);
        float sum = 0.0f;
        float p1 = -1.0f, p2 = -1.0f;
        int   i1 = 0,     i2 = 0;
        for (int e = 0; e < N_EXP; ++e) {
            const float pe = __expf(lg[e] - mx);
            sum += pe;
            if (pe > p1)      { p2 = p1; i2 = i1; p1 = pe; i1 = e; }
            else if (pe > p2) { p2 = pe; i2 = e; }
        }
        p1 /= sum; p2 /= sum;
        const float norm = 1.0f / (p1 + p2 + 1e-8f);
        const int t = m0 + tid;
        out_idx[2 * t]     = (float)i1;
        out_idx[2 * t + 1] = (float)i2;
        out_p[2 * t]       = p1 * norm;
        out_p[2 * t + 1]   = p2 * norm;
        atomicAdd(&usage[i1], 1.0f);
        atomicAdd(&usage[i2], 1.0f);
    }
}

// ---------------------------------------------------------------------------
// Launch
// ---------------------------------------------------------------------------
extern "C" void kernel_launch(void* const* d_in, const int* in_sizes, int n_in,
                              void* d_out, int out_size, void* d_ws, size_t ws_size,
                              hipStream_t stream) {
    (void)in_sizes; (void)n_in; (void)out_size; (void)ws_size;

    const float* x     = (const float*)d_in[0];
    const float* ln_g  = (const float*)d_in[1];
    const float* ln_b  = (const float*)d_in[2];
    const float* w1    = (const float*)d_in[3];
    const float* b1    = (const float*)d_in[4];
    const float* w2    = (const float*)d_in[5];
    const float* b2    = (const float*)d_in[6];
    const float* w3    = (const float*)d_in[7];
    const float* b3    = (const float*)d_in[8];
    const float* ebias = (const float*)d_in[9];
    // d_in[10] = top_k (== 2), compile-time assumed

    // workspace layout (bytes)
    char* ws = (char*)d_ws;
    bf16_t* w1g = (bf16_t*)(ws);                      // 16,777,216
    bf16_t* w2g = (bf16_t*)(ws + 16777216u);          // 16,777,216
    bf16_t* w3b = (bf16_t*)(ws + 33554432u);          //    524,288
    bf16_t* xnb = (bf16_t*)(ws + 34078720u);          // 67,108,864
    float*  b1f = (float*) (ws + 101187584u);         //     16,384
    float*  b2f = (float*) (ws + 101203968u);         //     16,384
    float*  b3f = (float*) (ws + 101220352u);         //        256

    float* out_idx = (float*)d_out;                   // [16384,2] indices
    float* out_p   = out_idx + 2 * N_TOKENS;          // [16384,2] probs
    float* out_use = out_idx + 4 * N_TOKENS;          // [64] usage

    prep_weights_kernel<<<2048, 256, 0, stream>>>(w1, w2, w3, ln_g, b3, ebias,
                                                  w1g, w2g, w3b, b3f, out_use);
    prep_bias_kernel<<<1024, 256, 0, stream>>>(w1, w2, ln_b, b1, b2, b1f, b2f);
    ln_norm_kernel<<<N_TOKENS / 8, 256, 0, stream>>>(x, xnb);
    moe_kernel<<<N_TOKENS / TILE_M, 512, 0, stream>>>(
        xnb, w1g, w2g, w3b, b1f, b2f, b3f, out_idx, out_p, out_use);
}
